// CrossAttentionLayer_75703093559626
// MI455X (gfx1250) — matrix operand order repaired and checked by reading, not compiled, not run on hardware
//
#include <hip/hip_runtime.h>
#include <hip/hip_bf16.h>

typedef __bf16 bf16_t;
typedef __attribute__((ext_vector_type(16))) __bf16 v16bf;
typedef __attribute__((ext_vector_type(8)))  float  v8f;

union FragU { v16bf v; unsigned u[8]; };

__device__ __forceinline__ unsigned short f32_to_bf16_bits(float f) {
  unsigned u = __builtin_bit_cast(unsigned, f);
  u += 0x7fffu + ((u >> 16) & 1u);          // round-to-nearest-even
  return (unsigned short)(u >> 16);
}
__device__ __forceinline__ unsigned pack2_bf16(float lo, float hi) {
  return (unsigned)f32_to_bf16_bits(lo) | ((unsigned)f32_to_bf16_bits(hi) << 16);
}

// ---------------------------------------------------------------------------
// A-fragment loader: 16x32 (MxK) bf16 fragment per ISA 7.12.2.
//   lane: M = lane&15 ; K-half offset = 8*(lane>>4)
//   VGPR i<4 : K = 2i, 2i+1 ; VGPR i>=4 : K = 2i+8, 2i+9   (+half offset)
// AM: 0 = row-major fp32 (elem (m,k) at m*lda+k)
//     1 = row-major bf16
//     2 = K-major bf16 (transposed: elem (m,k) at k*lda+m)  -- for attn^T
// ---------------------------------------------------------------------------
template <int AM>
__device__ __forceinline__ v16bf load_A(const void* __restrict__ Ab, int lda,
                                        int mBase, int kBase, int lane) {
  const int m    = mBase + (lane & 15);
  const int hoff = (lane >> 4) * 8;
  FragU fa;
#pragma unroll
  for (int i = 0; i < 8; ++i) {
    const int kk = kBase + hoff + ((i < 4) ? (2 * i) : (2 * i + 8));
    if (AM == 0) {
      const float* p = (const float*)Ab + (size_t)m * lda + kk;
      fa.u[i] = pack2_bf16(p[0], p[1]);
    } else if (AM == 1) {
      fa.u[i] = *(const unsigned*)((const unsigned short*)Ab + (size_t)m * lda + kk);
    } else {
      const unsigned short* p = (const unsigned short*)Ab;
      const unsigned lo = p[(size_t)kk * lda + m];
      const unsigned hi = p[(size_t)(kk + 1) * lda + m];
      fa.u[i] = lo | (hi << 16);
    }
  }
  return fa.v;
}

// ---------------------------------------------------------------------------
// B-fragment loader: 32x16 (KxN) bf16 fragment per ISA 7.12.2.
//   lane: N = lane&15 ; K-half offset = 8*(lane>>4)
//   VGPR j<4 : K = 2j, 2j+1 ; VGPR j>=4 : K = 2j+8, 2j+9   (+half offset)
// BM: 0 = N-major fp32  (elem (k,n) at n*ldb+k)  -- weight W (computes X*W^T)
//     1 = K-major bf16  (elem (k,n) at k*ldb+n)  -- V matrix
//     2 = K-major fp32                            -- vision matrix
//     3 = N-major bf16  (elem (k,n) at n*ldb+k)  -- K matrix (computes Q*K^T)
// ---------------------------------------------------------------------------
template <int BM>
__device__ __forceinline__ v16bf load_B(const void* __restrict__ Bb, int ldb,
                                        int kBase, int nBase, int lane) {
  const int n    = nBase + (lane & 15);
  const int hoff = (lane >> 4) * 8;
  FragU fb;
#pragma unroll
  for (int j = 0; j < 8; ++j) {
    const int kk = kBase + hoff + ((j < 4) ? (2 * j) : (2 * j + 8));
    if (BM == 0) {
      const float* p = (const float*)Bb + (size_t)n * ldb + kk;
      fb.u[j] = pack2_bf16(p[0], p[1]);
    } else if (BM == 1) {
      const unsigned short* p = (const unsigned short*)Bb;
      const unsigned lo = p[(size_t)kk * ldb + n];
      const unsigned hi = p[(size_t)(kk + 1) * ldb + n];
      fb.u[j] = lo | (hi << 16);
    } else if (BM == 2) {
      const float* p = (const float*)Bb;
      fb.u[j] = pack2_bf16(p[(size_t)kk * ldb + n], p[(size_t)(kk + 1) * ldb + n]);
    } else {
      const unsigned short* p = (const unsigned short*)Bb + (size_t)n * ldb + kk;
      fb.u[j] = *(const unsigned*)p;
    }
  }
  return fb.v;
}

__device__ __forceinline__ v8f wmma_bf16(v16bf a, v16bf b, v8f c) {
  return __builtin_amdgcn_wmma_f32_16x16x32_bf16(false, a, false, b, (short)0, c,
                                                 false, false);
}

// ---------------------------------------------------------------------------
// Generic wave-tile GEMM, ping-pong double-buffered (K must be a multiple of
// 64). Block = 128 threads (4 waves stacked in M); each wave computes a
// 32(M) x 64(N) tile: 2 A-frags x 4 B-frags -> 8 WMMAs per K-step of 32.
// Two fragment banks alternate: while bank X feeds the WMMAs, bank Y's
// global_load_b128 clauses are in flight -- no register rotation copies.
// ---------------------------------------------------------------------------
template <int AM, int BM, bool BIAS, bool OUTBF16>
__global__ __launch_bounds__(128) void gemm_wmma(
    const void* __restrict__ A, const void* __restrict__ B,
    const float* __restrict__ bias, void* __restrict__ C, int K, int lda,
    int ldb, int ldc, long long sA, long long sB, long long sC) {
  const int lane = threadIdx.x & 31;
  const int wave = threadIdx.x >> 5;
  const int z    = blockIdx.z;
  const int mBase = blockIdx.x * 128 + wave * 32;
  const int nBase = blockIdx.y * 64;

  const void* Ab = (AM == 0)
      ? (const void*)((const float*)A + (size_t)z * sA)
      : (const void*)((const unsigned short*)A + (size_t)z * sA);
  const void* Bb = (BM == 0 || BM == 2)
      ? (const void*)((const float*)B + (size_t)z * sB)
      : (const void*)((const unsigned short*)B + (size_t)z * sB);

  v8f acc[2][4];
#pragma unroll
  for (int i = 0; i < 2; ++i)
#pragma unroll
    for (int j = 0; j < 4; ++j) acc[i][j] = {};

  v16bf a0[2], b0[4], a1[2], b1[4];
#pragma unroll
  for (int i = 0; i < 2; ++i) a0[i] = load_A<AM>(Ab, lda, mBase + 16 * i, 0, lane);
#pragma unroll
  for (int j = 0; j < 4; ++j) b0[j] = load_B<BM>(Bb, ldb, 0, nBase + 16 * j, lane);

  for (int k0 = 0; k0 < K; k0 += 64) {
    // Prefetch bank 1 @ k0+32 (always valid: K % 64 == 0).
#pragma unroll
    for (int i = 0; i < 2; ++i)
      a1[i] = load_A<AM>(Ab, lda, mBase + 16 * i, k0 + 32, lane);
#pragma unroll
    for (int j = 0; j < 4; ++j)
      b1[j] = load_B<BM>(Bb, ldb, k0 + 32, nBase + 16 * j, lane);
    // Compute on bank 0.
#pragma unroll
    for (int i = 0; i < 2; ++i)
#pragma unroll
      for (int j = 0; j < 4; ++j) acc[i][j] = wmma_bf16(a0[i], b0[j], acc[i][j]);
    // Prefetch bank 0 @ k0+64 (uniform condition, skipped on last iter).
    if (k0 + 64 < K) {
#pragma unroll
      for (int i = 0; i < 2; ++i)
        a0[i] = load_A<AM>(Ab, lda, mBase + 16 * i, k0 + 64, lane);
#pragma unroll
      for (int j = 0; j < 4; ++j)
        b0[j] = load_B<BM>(Bb, ldb, k0 + 64, nBase + 16 * j, lane);
    }
    // Compute on bank 1.
#pragma unroll
    for (int i = 0; i < 2; ++i)
#pragma unroll
      for (int j = 0; j < 4; ++j) acc[i][j] = wmma_bf16(a1[i], b1[j], acc[i][j]);
  }

  const int half = lane >> 4;
  const int n0   = lane & 15;
#pragma unroll
  for (int i = 0; i < 2; ++i) {
#pragma unroll
    for (int j = 0; j < 4; ++j) {
#pragma unroll
      for (int r = 0; r < 8; ++r) {
        const int m = mBase + 16 * i + r + 8 * half;
        const int n = nBase + 16 * j + n0;
        float v = acc[i][j][r];
        if (BIAS) v += bias[n];
        if (OUTBF16) {
          ((unsigned short*)C + (size_t)z * sC)[(size_t)m * ldc + n] =
              f32_to_bf16_bits(v);
        } else {
          ((float*)C + (size_t)z * sC)[(size_t)m * ldc + n] = v;
        }
      }
    }
  }
}

// ---------------------------------------------------------------------------
// Fused scores + softmax: each wave owns 16 query rows. Raw f32 score strip
// (16 x St = 64 KB) lives in LDS (CDNA5: 320 KB/WGP, 128 KB per 2-wave block).
// Score loop: 4 t-tiles per Q fragment, ping-pong double-buffered over K
// (D % 64 == 0). Row stats reduced across the 16-lane halves with shfl_xor
// (C-layout rows m = r + 8*(lane>>4) live entirely within one half-wave).
// ---------------------------------------------------------------------------
__global__ __launch_bounds__(64) void attn_softmax_kernel(
    const unsigned short* __restrict__ Q, const unsigned short* __restrict__ Km,
    unsigned short* __restrict__ P, int Sv, int St, int D, float scale) {
  extern __shared__ float smem[];
  const int lane = threadIdx.x & 31;
  const int wave = threadIdx.x >> 5;
  const int half = lane >> 4;
  const int n0   = lane & 15;
  const int b    = blockIdx.y;
  const int sBase = blockIdx.x * 32 + wave * 16;

  const unsigned short* Qb = Q + (size_t)b * Sv * D;
  const unsigned short* Kb = Km + (size_t)b * St * D;
  unsigned short* Pb = P + (size_t)b * Sv * St + (size_t)sBase * St;
  float* strip = smem + (size_t)wave * 16 * St;

  float pmax[8];
#pragma unroll
  for (int r = 0; r < 8; ++r) pmax[r] = -3.0e38f;

  // Pass 1: scores = (Q K^T) * scale, strip -> LDS, running row max.
  for (int t = 0; t < St / 16; t += 4) {
    v8f acc[4];
#pragma unroll
    for (int j = 0; j < 4; ++j) acc[j] = {};

    v16bf a0 = load_A<1>(Qb, D, sBase, 0, lane);
    v16bf bf0[4], a1, bf1[4];
#pragma unroll
    for (int j = 0; j < 4; ++j)
      bf0[j] = load_B<3>(Kb, D, 0, (t + j) * 16, lane);

    for (int k0 = 0; k0 < D; k0 += 64) {
      a1 = load_A<1>(Qb, D, sBase, k0 + 32, lane);
#pragma unroll
      for (int j = 0; j < 4; ++j)
        bf1[j] = load_B<3>(Kb, D, k0 + 32, (t + j) * 16, lane);
#pragma unroll
      for (int j = 0; j < 4; ++j) acc[j] = wmma_bf16(a0, bf0[j], acc[j]);
      if (k0 + 64 < D) {
        a0 = load_A<1>(Qb, D, sBase, k0 + 64, lane);
#pragma unroll
        for (int j = 0; j < 4; ++j)
          bf0[j] = load_B<3>(Kb, D, k0 + 64, (t + j) * 16, lane);
      }
#pragma unroll
      for (int j = 0; j < 4; ++j) acc[j] = wmma_bf16(a1, bf1[j], acc[j]);
    }
#pragma unroll
    for (int j = 0; j < 4; ++j) {
#pragma unroll
      for (int r = 0; r < 8; ++r) {
        const float s = acc[j][r] * scale;
        pmax[r] = fmaxf(pmax[r], s);
        strip[(size_t)(r + 8 * half) * St + (t + j) * 16 + n0] = s;
      }
    }
  }
#pragma unroll
  for (int r = 0; r < 8; ++r)
#pragma unroll
    for (int off = 8; off >= 1; off >>= 1)
      pmax[r] = fmaxf(pmax[r], __shfl_xor(pmax[r], off, 32));

  // Pass 2: exp(s - max) in LDS, row sums. Each lane owns 64 contiguous cols.
  const int colBase = n0 * (St / 16);
  float psum[8];
#pragma unroll
  for (int r = 0; r < 8; ++r) {
    const int row = r + 8 * half;
    const float mx = pmax[r];
    float s = 0.f;
    for (int c = 0; c < St / 16; ++c) {
      const float e = __expf(strip[(size_t)row * St + colBase + c] - mx);
      strip[(size_t)row * St + colBase + c] = e;
      s += e;
    }
    psum[r] = s;
  }
#pragma unroll
  for (int r = 0; r < 8; ++r)
#pragma unroll
    for (int off = 8; off >= 1; off >>= 1)
      psum[r] += __shfl_xor(psum[r], off, 32);

  // Pass 3: normalize, convert to bf16, store probabilities.
#pragma unroll
  for (int r = 0; r < 8; ++r) {
    const int row = r + 8 * half;
    const float inv = 1.0f / psum[r];
    for (int c = 0; c < St / 16; c += 2) {
      const float e0 = strip[(size_t)row * St + colBase + c] * inv;
      const float e1 = strip[(size_t)row * St + colBase + c + 1] * inv;
      *(unsigned*)(Pb + (size_t)row * St + colBase + c) = pack2_bf16(e0, e1);
    }
  }
}

extern "C" void kernel_launch(void* const* d_in, const int* in_sizes, int n_in,
                              void* d_out, int out_size, void* d_ws,
                              size_t ws_size, hipStream_t stream) {
  (void)in_sizes; (void)n_in; (void)out_size; (void)ws_size;
  const int B = 16, Sv = 1024, St = 1024, D = 1024;
  const float scale = 1.0f / 32.0f;  // 1/sqrt(1024)

  const float* vision = (const float*)d_in[0];
  const float* text   = (const float*)d_in[1];
  const float* Wq = (const float*)d_in[2];
  const float* bq = (const float*)d_in[3];
  const float* Wk = (const float*)d_in[4];
  const float* bk = (const float*)d_in[5];
  const float* Wv = (const float*)d_in[6];
  const float* bv = (const float*)d_in[7];

  unsigned short* Qw = (unsigned short*)d_ws;                   // bf16 [B,Sv,D]
  unsigned short* Kw = Qw + (size_t)B * Sv * D;                 // bf16 [B,St,D]
  unsigned short* Vw = Kw + (size_t)B * St * D;                 // bf16 [B,St,D]
  unsigned short* Pw = Vw + (size_t)B * St * D;                 // bf16 [B,Sv,St]

  float* cross_vision = (float*)d_out;
  float* cross_text   = cross_vision + (size_t)B * Sv * D;

  // 1) Projections: X[BS,D](f32) x W^T(f32) + b -> bf16. Flatten batch (M=BS).
  {
    dim3 grid((B * Sv) / 128, D / 64, 1);
    gemm_wmma<0, 0, true, true><<<grid, 128, 0, stream>>>(
        vision, Wq, bq, Qw, D, D, D, D, 0, 0, 0);
    gemm_wmma<0, 0, true, true><<<grid, 128, 0, stream>>>(
        text, Wk, bk, Kw, D, D, D, D, 0, 0, 0);
    gemm_wmma<0, 0, true, true><<<grid, 128, 0, stream>>>(
        text, Wv, bv, Vw, D, D, D, D, 0, 0, 0);
  }

  // 2) Fused scores + softmax -> bf16 probabilities P.
  {
    dim3 grid(Sv / 32, B);
    const size_t lds = 2u * 16u * (size_t)St * sizeof(float);  // 128 KB
    attn_softmax_kernel<<<grid, 64, lds, stream>>>(Qw, Kw, Pw, Sv, St, D, scale);
  }

  // 3) cross_vision = P x V   (bf16 x bf16 -> f32), batched over z.
  {
    dim3 grid(Sv / 128, D / 64, B);
    gemm_wmma<1, 1, false, false><<<grid, 128, 0, stream>>>(
        Pw, Vw, nullptr, cross_vision, St, St, D, D,
        (long long)Sv * St, (long long)St * D, (long long)Sv * D);
  }

  // 4) cross_text = P^T x vision (bf16^T x f32 -> f32), batched over z.
  {
    dim3 grid(St / 128, D / 64, B);
    gemm_wmma<2, 2, false, false><<<grid, 128, 0, stream>>>(
        Pw, vision, nullptr, cross_text, Sv, St, D, D,
        (long long)Sv * St, (long long)Sv * D, (long long)St * D);
  }
}
